// WayFormerEncoder_90958817394913
// MI455X (gfx1250) — matrix operand order repaired and checked by reading, not compiled, not run on hardware
//
#include <hip/hip_runtime.h>

// ---------------- problem constants ----------------
#define Bn   16
#define Sn   128
#define Tn   64
#define Hn   128
#define NHn  8
#define DHn  16
#define FFn  256
#define NLn  4
#define NTOK (Bn * Sn * Tn)          // 131072 tokens
#define SCALE_ATT 0.25f              // 1/sqrt(16)

// ---------------- WMMA / vector types ----------------
typedef __attribute__((ext_vector_type(16))) __bf16        v16bf;
typedef __attribute__((ext_vector_type(8)))  float         v8f;
typedef __attribute__((ext_vector_type(4)))  unsigned int  u32x4;
typedef __attribute__((ext_vector_type(8)))  int           i32x8;
typedef __attribute__((ext_vector_type(4)))  int           i32x4;

union FragU { v16bf v; u32x4 q[2]; };

// A-matrix 16x32 bf16 fragment (lane<16: row=lane, K=0..7 & 16..23; lane>=16: row=lane-16, K=8..15 & 24..31)
__device__ __forceinline__ v16bf load_frag_a(const __bf16* base, int ld, int row0, int k0, int lane) {
    int r  = row0 + (lane & 15);
    int kb = k0 + ((lane >> 4) << 3);
    const __bf16* p = base + (size_t)r * ld + kb;
    FragU f;
    f.q[0] = *(const u32x4*)(p);        // K = kb .. kb+7
    f.q[1] = *(const u32x4*)(p + 16);   // K = kb+16 .. kb+23
    return f.v;
}

// B-matrix 32x16 bf16 fragment from K-major storage Wt[col][k] (lane<16: col=lane, K=0..7 & 16..23; lane>=16: col=lane-16, K=8..15 & 24..31)
__device__ __forceinline__ v16bf load_frag_b(const __bf16* base, int ld, int col0, int k0, int lane) {
    int c  = col0 + (lane & 15);
    int kb = k0 + ((lane >> 4) << 3);
    const __bf16* p = base + (size_t)c * ld + kb;
    FragU f;
    f.q[0] = *(const u32x4*)(p);        // K = kb .. kb+7
    f.q[1] = *(const u32x4*)(p + 16);   // K = kb+16 .. kb+23
    return f.v;
}

__device__ __forceinline__ v8f wmma_bf16(v16bf a, v16bf b, v8f c) {
    return __builtin_amdgcn_wmma_f32_16x16x32_bf16(false, a, false, b, (short)0, c, false, false);
}

// ---------------- TDM: async 2-D tile load global -> LDS ----------------
// Loads a [rows x K] bf16 tile (row-major, contiguous rows, stride = K elements)
// starting at `gsrc` into LDS at `lds`. Issued by the calling wave; caller must
// barrier (s_wait_tensorcnt is done here). Descriptor per CDNA5 ISA 8.3/8.4.
__device__ __forceinline__ void tdm_load_tile_b16(void* lds, const void* gsrc,
                                                  int K, int rows, int tensor_rows) {
#if __has_builtin(__builtin_amdgcn_tensor_load_to_lds)
    unsigned long long ga = (unsigned long long)gsrc;
    unsigned ldsa = (unsigned)(unsigned long long)lds;   // flat addr low 32 bits == LDS offset mapping
    u32x4 g0;
    g0.x = 1u;                                           // count=1 (valid), user descriptor
    g0.y = ldsa;                                         // lds_addr
    g0.z = (unsigned)(ga & 0xFFFFFFFFu);                 // global_addr[31:0]
    g0.w = (unsigned)((ga >> 32) & 0x01FFFFFFu) | (2u << 30); // global_addr[56:32] | type=2 (image)
    i32x8 g1;
    g1[0] = (1 << 16);                                   // workgroup_mask=0, data_size=1 (2 bytes)
    g1[1] = (int)(((unsigned)K & 0xFFFFu) << 16);        // tensor_dim0[15:0] in bits[63:48]
    g1[2] = (int)((((unsigned)K >> 16) & 0xFFFFu) |      // tensor_dim0[31:16]
                  (((unsigned)tensor_rows & 0xFFFFu) << 16)); // tensor_dim1[15:0]
    g1[3] = (int)((((unsigned)tensor_rows >> 16) & 0xFFFFu) | // tensor_dim1[31:16]
                  (((unsigned)K & 0xFFFFu) << 16));      // tile_dim0 = K
    g1[4] = (int)((unsigned)rows & 0xFFFFu);             // tile_dim1 = rows, tile_dim2 = 0
    g1[5] = K;                                           // tensor_dim0_stride[31:0]
    g1[6] = 0;                                           // stride hi / dim1_stride lo
    g1[7] = 0;
    i32x4 g2 = {0, 0, 0, 0}, g3 = {0, 0, 0, 0};
#if __clang_major__ >= 23
    i32x8 g4 = {0, 0, 0, 0, 0, 0, 0, 0};
    __builtin_amdgcn_tensor_load_to_lds(g0, g1, g2, g3, g4, 0);
#else
    __builtin_amdgcn_tensor_load_to_lds(g0, g1, g2, g3, 0);
#endif
    __builtin_amdgcn_s_wait_tensorcnt(0);
#endif
}

__device__ __forceinline__ constexpr bool tdm_available() {
#if __has_builtin(__builtin_amdgcn_tensor_load_to_lds)
    return true;
#else
    return false;
#endif
}

// ---------------- weight convert + transpose: W[K][M] fp32 -> Wt[M][K] bf16 ----------------
__global__ void wconv_kernel(const float* __restrict__ W, __bf16* __restrict__ Wt, int K, int M) {
    int idx = blockIdx.x * 256 + threadIdx.x;
    if (idx >= K * M) return;
    int k = idx / M, m = idx - k * M;
    Wt[(size_t)m * K + k] = (__bf16)W[idx];
}

// ---------------- masked input build: x = valid ? poly : -1, as bf16 ----------------
__global__ void build_input_kernel(const float* __restrict__ poly, const int* __restrict__ lengths,
                                   __bf16* __restrict__ Xb) {
    int idx = blockIdx.x * 256 + threadIdx.x;          // over B*S*T*H
    int q   = idx >> 7;
    int t   = q & (Tn - 1);
    int bs  = q >> 6;
    float v = (t < lengths[bs]) ? poly[idx] : -1.0f;
    Xb[idx] = (__bf16)v;
}

// ---------------- generic WMMA GEMM: C[N][M] = A[N][K] @ Wt[M][K]^T + bias ----------------
// block = 256 threads (8 waves). Block computes 128 rows x 64 cols.
// Each wave: 16 rows x 64 cols = 4 accumulators; one A fragment feeds 4 WMMAs per k-step.
__global__ void gemm_kernel(const __bf16* __restrict__ A, const __bf16* __restrict__ Wt,
                            const float* __restrict__ bias, int N, int K, int M,
                            float* __restrict__ Cf, __bf16* __restrict__ Cb, int relu) {
    __shared__ __bf16 Bs[64 * 256];                    // up to 32KB: 64 cols x K, K-major
    const int tid = threadIdx.x, lane = tid & 31, wave = tid >> 5;
    const int c0 = blockIdx.x * 64;
    const int r0 = blockIdx.y * 128 + wave * 16;

    // stage 64 columns of Wt (64*K contiguous bf16) into LDS via TDM (async DMA), else copy
    if (tdm_available()) {
        if (wave == 0)
            tdm_load_tile_b16(Bs, Wt + (size_t)c0 * K, K, 64, M);
    } else {
        const u32x4* src = (const u32x4*)(Wt + (size_t)c0 * K);
        u32x4* dst = (u32x4*)Bs;
        const int nvec = (64 * K) >> 3;
        for (int i = tid; i < nvec; i += 256) dst[i] = src[i];
    }
    __syncthreads();

    v8f acc0 = {}, acc1 = {}, acc2 = {}, acc3 = {};
    for (int k0 = 0; k0 < K; k0 += 32) {
        if (k0 + 32 < K)
            __builtin_prefetch(A + (size_t)(r0 + (lane & 15)) * K + k0 + 32, 0, 3);
        v16bf a = load_frag_a(A, K, r0, k0, lane);
        acc0 = wmma_bf16(a, load_frag_b(Bs, K,  0, k0, lane), acc0);
        acc1 = wmma_bf16(a, load_frag_b(Bs, K, 16, k0, lane), acc1);
        acc2 = wmma_bf16(a, load_frag_b(Bs, K, 32, k0, lane), acc2);
        acc3 = wmma_bf16(a, load_frag_b(Bs, K, 48, k0, lane), acc3);
    }

    v8f accs[4] = {acc0, acc1, acc2, acc3};
#pragma unroll
    for (int ct = 0; ct < 4; ++ct) {
        const int c = c0 + ct * 16 + (lane & 15);
        const float bv = bias ? bias[c] : 0.0f;
#pragma unroll
        for (int r = 0; r < 8; ++r) {
            int row = r0 + r + ((lane >> 4) << 3);
            float v = accs[ct][r] + bv;
            if (relu && v < 0.0f) v = 0.0f;
            if (Cf) Cf[(size_t)row * M + c] = v;
            if (Cb) Cb[(size_t)row * M + c] = (__bf16)v;
        }
    }
}

// ---------------- fused attention per (sequence, head) ----------------
// grid = (nseq, NH), block = 256. L = 64 (temporal) or 128 (spatial).
__global__ void attn_kernel(const __bf16* __restrict__ Qb, const __bf16* __restrict__ Kb,
                            const __bf16* __restrict__ Vb, const int* __restrict__ lengths,
                            __bf16* __restrict__ Ob, int L, int spatial) {
    __shared__ __bf16 Qh[128 * 32];    // [tok][32] zero-padded K
    __shared__ __bf16 Kh[128 * 32];
    __shared__ __bf16 VhT[16 * 128];   // [d][tok]
    __shared__ float  Sc[64 * 128];    // score panel: 64 query rows x L
    __shared__ int    validS[128];

    const int n = blockIdx.x, hh = blockIdx.y;
    const int tid = threadIdx.x, lane = tid & 31, wave = tid >> 5;

    for (int t = tid; t < L; t += 256) {
        int ok;
        if (!spatial) ok = (t < lengths[n]);                       // n = b*S + s
        else { int b = n >> 6; int tp = n & 63; ok = (tp < lengths[b * Sn + t]); }
        validS[t] = ok;
    }
    for (int i = tid; i < L * DHn; i += 256) {
        int t = i >> 4, d = i & 15;
        size_t row = (size_t)n * L + t;
        Qh[t * 32 + d]       = Qb[row * Hn + hh * DHn + d];
        Qh[t * 32 + 16 + d]  = (__bf16)0.0f;
        Kh[t * 32 + d]       = Kb[row * Hn + hh * DHn + d];
        Kh[t * 32 + 16 + d]  = (__bf16)0.0f;
        VhT[d * L + t]       = Vb[row * Hn + hh * DHn + d];
    }
    __syncthreads();

    const int nkt = L >> 4;                    // key tiles
    for (int qp = 0; qp < L; qp += 64) {       // 64-row query panels
        // ---- scores: one WMMA per 16x16 tile (K padded 16->32) ----
        for (int tile = wave; tile < 4 * nkt; tile += 8) {
            int qi = tile / nkt, ki = tile - qi * nkt;
            v16bf a = load_frag_a(Qh, 32, qp + qi * 16, 0, lane);
            v16bf b = load_frag_b(Kh, 32, ki * 16, 0, lane);
            v8f c = {};
            c = wmma_bf16(a, b, c);
            int col = ki * 16 + (lane & 15);
#pragma unroll
            for (int r = 0; r < 8; ++r) {
                int ql = qi * 16 + r + ((lane >> 4) << 3);
                int qg = qp + ql;
                float v = (validS[qg] && validS[col]) ? c[r] * SCALE_ATT : -1e9f;
                Sc[ql * L + col] = v;
            }
        }
        __syncthreads();

        // ---- softmax per row + in-place fp32 -> bf16 conversion ----
        if (tid < 64) {
            float* rowp = &Sc[tid * L];
            float m = -3.0e38f;
            for (int k = 0; k < L; ++k) m = fmaxf(m, rowp[k]);
            float s = 0.0f;
            for (int k = 0; k < L; ++k) { float e = __expf(rowp[k] - m); rowp[k] = e; s += e; }
            float inv = 1.0f / s;
            __bf16* ab = (__bf16*)rowp;                 // safe: byte 2k < 4k when reading float k
            for (int k = 0; k < L; ++k) ab[k] = (__bf16)(rowp[k] * inv);
        }
        __syncthreads();

        // ---- O panel = A(64 x L) @ V(L x 16), A rows packed with bf16 ld = 2L ----
        for (int qi = wave; qi < 4; qi += 8) {
            v8f c = {};
            for (int k0 = 0; k0 < L; k0 += 32) {
                v16bf a = load_frag_a((const __bf16*)Sc, 2 * L, qi * 16, k0, lane);
                v16bf b = load_frag_b(VhT, L, 0, k0, lane);
                c = wmma_bf16(a, b, c);
            }
            int d = lane & 15;
#pragma unroll
            for (int r = 0; r < 8; ++r) {
                int qg = qp + qi * 16 + r + ((lane >> 4) << 3);
                size_t row = (size_t)n * L + qg;
                Ob[row * Hn + hh * DHn + d] = (__bf16)c[r];
            }
        }
        __syncthreads();
    }
}

// ---------------- LN helpers ----------------
__global__ void ln_relu_pos_kernel(const float* __restrict__ Y,
                                   const float* __restrict__ s, const float* __restrict__ b,
                                   const float* __restrict__ pos,
                                   float* __restrict__ Xf, __bf16* __restrict__ Xb) {
    __shared__ float red[128];
    const int row = blockIdx.x, h = threadIdx.x;
    float x = Y[(size_t)row * Hn + h];
    red[h] = x; __syncthreads();
    for (int o = 64; o > 0; o >>= 1) { if (h < o) red[h] += red[h + o]; __syncthreads(); }
    float mean = red[0] * (1.0f / Hn); __syncthreads();
    float d = x - mean;
    red[h] = d * d; __syncthreads();
    for (int o = 64; o > 0; o >>= 1) { if (h < o) red[h] += red[h + o]; __syncthreads(); }
    float rs = rsqrtf(red[0] * (1.0f / Hn) + 1e-5f);
    float v = d * rs * s[h] + b[h];
    if (v < 0.0f) v = 0.0f;
    v += pos[(row & (Tn - 1)) * Hn + h];
    Xf[(size_t)row * Hn + h] = v;
    Xb[(size_t)row * Hn + h] = (__bf16)v;
}

__global__ void resid_ln_kernel(float* __restrict__ Xf, const float* __restrict__ F,
                                const float* __restrict__ s, const float* __restrict__ b,
                                __bf16* __restrict__ Xb) {
    __shared__ float red[128];
    const int row = blockIdx.x, h = threadIdx.x;
    float x = Xf[(size_t)row * Hn + h] + F[(size_t)row * Hn + h];
    red[h] = x; __syncthreads();
    for (int o = 64; o > 0; o >>= 1) { if (h < o) red[h] += red[h + o]; __syncthreads(); }
    float mean = red[0] * (1.0f / Hn); __syncthreads();
    float d = x - mean;
    red[h] = d * d; __syncthreads();
    for (int o = 64; o > 0; o >>= 1) { if (h < o) red[h] += red[h + o]; __syncthreads(); }
    float rs = rsqrtf(red[0] * (1.0f / Hn) + 1e-5f);
    float v = d * rs * s[h] + b[h];
    Xf[(size_t)row * Hn + h] = v;
    Xb[(size_t)row * Hn + h] = (__bf16)v;
}

// ---------------- temporal [B*S][T][H] -> spatial [B*T][S][H] ----------------
__global__ void transpose_kernel(const float* __restrict__ src,
                                 float* __restrict__ dstF, __bf16* __restrict__ dstB) {
    int idx = blockIdx.x * 256 + threadIdx.x;          // dst linear over B*T*S*H
    int h  = idx & (Hn - 1);
    int q  = idx >> 7;
    int sx = q & (Sn - 1);
    int q2 = q >> 7;
    int t  = q2 & (Tn - 1);
    int b  = q2 >> 6;
    float v = src[(((size_t)(b * Sn + sx) * Tn + t) * Hn) + h];
    dstF[idx] = v;
    dstB[idx] = (__bf16)v;
}

// ---------------- final gather: out[b][t][h] = X[(b*T+0)][s=t][h] ----------------
__global__ void gather_kernel(const float* __restrict__ X, float* __restrict__ out) {
    int idx = blockIdx.x * 256 + threadIdx.x;          // over B*T*H
    int h = idx & (Hn - 1);
    int t = (idx >> 7) & (Tn - 1);
    int b = idx >> 13;
    out[idx] = X[(((size_t)(b * Tn) * Sn + t) * Hn) + h];
}

// ---------------- host orchestration ----------------
extern "C" void kernel_launch(void* const* d_in, const int* in_sizes, int n_in,
                              void* d_out, int out_size, void* d_ws, size_t ws_size,
                              hipStream_t stream) {
    const float* poly   = (const float*)d_in[0];
    const int*   lens   = (const int*)  d_in[1];
    const float* proj_w = (const float*)d_in[2];
    const float* proj_b = (const float*)d_in[3];
    const float* pln_s  = (const float*)d_in[4];
    const float* pln_b  = (const float*)d_in[5];
    const float* pos    = (const float*)d_in[6];
    const float* Wq = (const float*)d_in[7],  *bq = (const float*)d_in[8];
    const float* Wk = (const float*)d_in[9],  *bk = (const float*)d_in[10];
    const float* Wv = (const float*)d_in[11], *bv = (const float*)d_in[12];
    const float* Wo = (const float*)d_in[13], *bo = (const float*)d_in[14];
    const float* l1s = (const float*)d_in[15], *l1b = (const float*)d_in[16];
    const float* W1 = (const float*)d_in[17], *b1 = (const float*)d_in[18];
    const float* W2 = (const float*)d_in[19], *b2 = (const float*)d_in[20];
    const float* l2s = (const float*)d_in[21], *l2b = (const float*)d_in[22];
    float* out = (float*)d_out;

    const size_t MB = 1024ull * 1024ull;
    char* ws = (char*)d_ws;
    float*  XfA = (float*) (ws);              // 64 MB
    __bf16* XbA = (__bf16*)(ws + 64  * MB);   // 32 MB
    float*  XfB = (float*) (ws + 96  * MB);   // 64 MB
    __bf16* XbB = (__bf16*)(ws + 160 * MB);   // 32 MB
    __bf16* Qb  = (__bf16*)(ws + 192 * MB);   // 32 MB
    __bf16* Kbuf= (__bf16*)(ws + 224 * MB);   // 32 MB
    __bf16* Vb  = (__bf16*)(ws + 256 * MB);   // 32 MB
    __bf16* Ob  = (__bf16*)(ws + 288 * MB);   // 32 MB
    __bf16* Hb  = (__bf16*)(ws + 320 * MB);   // 64 MB (FFN hidden bf16)
    float*  Sf  = (float*) (ws + 384 * MB);   // 64 MB (fp32 GEMM scratch)
    __bf16* Wb  = (__bf16*)(ws + 448 * MB);   // bf16 transposed weights

    __bf16* projT = Wb;                                   // 128*128
    const size_t HH = (size_t)Hn * Hn, HF = (size_t)Hn * FFn;
    const size_t LSTRIDE = 4 * HH + 2 * HF;
    auto WqT = [&](int l){ return projT + HH + l * LSTRIDE + 0 * HH; };
    auto WkT = [&](int l){ return projT + HH + l * LSTRIDE + 1 * HH; };
    auto WvT = [&](int l){ return projT + HH + l * LSTRIDE + 2 * HH; };
    auto WoT = [&](int l){ return projT + HH + l * LSTRIDE + 3 * HH; };
    auto W1T = [&](int l){ return projT + HH + l * LSTRIDE + 4 * HH; };          // [FF][H]
    auto W2T = [&](int l){ return projT + HH + l * LSTRIDE + 4 * HH + HF; };     // [H][FF]

    // ---- convert all weights to bf16 [M][K] ----
    wconv_kernel<<<(int)((HH + 255) / 256), 256, 0, stream>>>(proj_w, projT, Hn, Hn);
    for (int l = 0; l < NLn; ++l) {
        wconv_kernel<<<(int)((HH + 255) / 256), 256, 0, stream>>>(Wq + l * HH, WqT(l), Hn, Hn);
        wconv_kernel<<<(int)((HH + 255) / 256), 256, 0, stream>>>(Wk + l * HH, WkT(l), Hn, Hn);
        wconv_kernel<<<(int)((HH + 255) / 256), 256, 0, stream>>>(Wv + l * HH, WvT(l), Hn, Hn);
        wconv_kernel<<<(int)((HH + 255) / 256), 256, 0, stream>>>(Wo + l * HH, WoT(l), Hn, Hn);
        wconv_kernel<<<(int)((HF + 255) / 256), 256, 0, stream>>>(W1 + l * HF, W1T(l), Hn, FFn);
        wconv_kernel<<<(int)((HF + 255) / 256), 256, 0, stream>>>(W2 + l * HF, W2T(l), FFn, Hn);
    }

    auto gemm = [&](const __bf16* A, const __bf16* Wt, const float* bias,
                    int N, int K, int M, float* Cf, __bf16* Cb, int relu) {
        dim3 grid(M / 64, N / 128);
        gemm_kernel<<<grid, 256, 0, stream>>>(A, Wt, bias, N, K, M, Cf, Cb, relu);
    };

    // ---- input: mask/pad(-1) -> bf16, proj GEMM, LN+ReLU+pos ----
    build_input_kernel<<<NTOK * Hn / 256, 256, 0, stream>>>(poly, lens, XbA);
    gemm(XbA, projT, proj_b, NTOK, Hn, Hn, Sf, nullptr, 0);
    ln_relu_pos_kernel<<<NTOK, 128, 0, stream>>>(Sf, pln_s, pln_b, pos, XfA, XbA);

    float*  Xf = XfA;
    __bf16* Xb = XbA;
    for (int l = 0; l < NLn; ++l) {
        const int L = (l < 2) ? Tn : Sn;
        const int nseq = NTOK / L;
        gemm(Xb, WqT(l), bq + l * Hn, NTOK, Hn, Hn, nullptr, Qb, 0);
        gemm(Xb, WkT(l), bk + l * Hn, NTOK, Hn, Hn, nullptr, Kbuf, 0);
        gemm(Xb, WvT(l), bv + l * Hn, NTOK, Hn, Hn, nullptr, Vb, 0);
        attn_kernel<<<dim3(nseq, NHn), 256, 0, stream>>>(Qb, Kbuf, Vb, lens, Ob, L, (l >= 2) ? 1 : 0);
        gemm(Ob, WoT(l), bo + l * Hn, NTOK, Hn, Hn, Sf, nullptr, 0);
        resid_ln_kernel<<<NTOK, 128, 0, stream>>>(Xf, Sf, l1s + l * Hn, l1b + l * Hn, Xb);
        gemm(Xb, W1T(l), b1 + l * FFn, NTOK, Hn, FFn, nullptr, Hb, 1);
        gemm(Hb, W2T(l), b2 + l * Hn, NTOK, FFn, Hn, Sf, nullptr, 0);
        resid_ln_kernel<<<NTOK, 128, 0, stream>>>(Xf, Sf, l2s + l * Hn, l2b + l * Hn, Xb);
        if (l == 1) {  // temporal -> spatial re-layout
            transpose_kernel<<<NTOK * Hn / 256, 256, 0, stream>>>(Xf, XfB, XbB);
            Xf = XfB; Xb = XbB;
        }
    }
    gather_kernel<<<(Bn * Tn * Hn) / 256, 256, 0, stream>>>(Xf, out);
}
